// HypE_77292231459654
// MI455X (gfx1250) — matrix operand order repaired and checked by reading, not compiled, not run on hardware
//
#include <hip/hip_runtime.h>
#include <hip/hip_bf16.h>

// ---------------- problem constants ----------------
#define BATCH   8192
#define EMB     400          // embedding dim (N of the GEMM)
#define WOUT    200          // e[:, ::2] length (K of the GEMM)
#define KPAD    224          // K padded to multiple of 32
#define KT      7            // KPAD / 32 k-tiles
#define NT      25           // EMB / 16 n-tiles
#define OUTCH   6
#define PITCH   232          // LDS row pitch in bf16 elements (mult of 8 -> 16B aligned rows)

typedef __attribute__((ext_vector_type(16))) __bf16        v16bf;
typedef __attribute__((ext_vector_type(8)))  float         v8f;
typedef __attribute__((ext_vector_type(4)))  unsigned int  v4u;

union AB16 { v4u q[2]; v16bf v; };   // two 16B chunks -> one 32B WMMA operand

__device__ __forceinline__ unsigned short f2bf(float f) {
    unsigned int u = __float_as_uint(f);
    // round-to-nearest-even bf16 truncation
    unsigned int r = (u + 0x7FFFu + ((u >> 16) & 1u)) >> 16;
    return (unsigned short)r;
}
__device__ __forceinline__ unsigned int pack2(float lo, float hi) {
    return (unsigned int)f2bf(lo) | ((unsigned int)f2bf(hi) << 16);
}

// ---------------------------------------------------------------------------
// Prep: W_eff^{(i)}[d,w] = sum_c (fc2_W[c,i]+fc2_b[c]) * fc_W[d, c*200+w]
// stored as bf16 pairs directly in the WMMA B-operand per-lane layout:
//   flat dword = (((i*KT+kt)*NT+dt)*32 + lane)*8 + j
//   lane: n=lane&15 (matrix column d), half=lane>>4; dword j: K = half*16+2j, +1
// ---------------------------------------------------------------------------
__global__ void hype_prep_kernel(const float* __restrict__ fc2W,
                                 const float* __restrict__ fc2b,
                                 const float* __restrict__ fcW,
                                 unsigned int* __restrict__ Bp)
{
    int t = blockIdx.x * blockDim.x + threadIdx.x;
    const int total = 3 * KT * NT * 32 * 8;
    if (t >= total) return;

    int j    = t & 7;
    int lane = (t >> 3) & 31;
    int g    = t >> 8;           // (i*KT+kt)*NT + dt
    int dt   = g % NT;
    int g2   = g / NT;
    int kt   = g2 % KT;
    int i    = g2 / KT;

    int n    = lane & 15;
    int half = lane >> 4;
    int d    = dt * 16 + n;
    int k0   = kt * 32 + half * 16 + 2 * j;   // w index of low element

    float kc[OUTCH];
#pragma unroll
    for (int c = 0; c < OUTCH; ++c)
        kc[c] = fc2W[c * 4 + i] + fc2b[c];    // fc2_W is (6,4) row-major

    float v0 = 0.f, v1 = 0.f;
    if (k0 < WOUT) {
#pragma unroll
        for (int c = 0; c < OUTCH; ++c)
            v0 += kc[c] * fcW[d * 1200 + c * WOUT + k0];
    }
    if (k0 + 1 < WOUT) {
#pragma unroll
        for (int c = 0; c < OUTCH; ++c)
            v1 += kc[c] * fcW[d * 1200 + c * WOUT + k0 + 1];
    }
    Bp[t] = pack2(v0, v1);
}

// ---------------------------------------------------------------------------
// Main: one block per 16-row batch tile; 8 waves split the 25 dim-tiles.
// ---------------------------------------------------------------------------
__global__ void __launch_bounds__(256)
hype_main_kernel(const int*   __restrict__ r_idx,
                 const int*   __restrict__ E_mat,
                 const float* __restrict__ ms,
                 const float* __restrict__ bs,
                 const float* __restrict__ E_emb,
                 const float* __restrict__ R_emb,
                 const float* __restrict__ fc_b,
                 const v4u*   __restrict__ Bp,
                 float*       __restrict__ out)
{
    __shared__ __align__(16) unsigned short smA[3 * 16 * PITCH]; // bf16 e_s tiles
    __shared__ float sOut[16];
    __shared__ float sMs[48], sBs[48];
    __shared__ int   sRidx[16];

    const int tid = threadIdx.x;
    const int b0  = blockIdx.x * 16;

    if (tid < 16) { sOut[tid] = 0.f; sRidx[tid] = r_idx[b0 + tid]; }
    if (tid < 48) {
        int m = tid / 3, i = tid % 3;
        sMs[i * 16 + m] = ms[(b0 + m) * 3 + i];
        sBs[i * 16 + m] = bs[(b0 + m) * 3 + i];
    }

    // Stage e_s = E_emb[idx][::2] -> bf16 in LDS. One float4 load -> one dword.
    for (int p = tid; p < 3 * 16 * 112; p += 256) {
        int pp = p % 112;            // float4 index within the row (2 e_s elems)
        int m  = (p / 112) & 15;
        int i  = p / (112 * 16);
        unsigned int val = 0u;
        if (pp < 100) {
            int eidx = E_mat[(b0 + m) * 3 + i];
            const float4 f = ((const float4*)(E_emb + (size_t)eidx * EMB))[pp];
            val = pack2(f.x, f.z);   // elements 2*pp and 2*pp+1 of e_s
        }
        *(unsigned int*)&smA[(i * 16 + m) * PITCH + 2 * pp] = val;
    }
    __syncthreads();

    const int lane = tid & 31;
    const int wv   = tid >> 5;
    const int n    = lane & 15;
    const int half = lane >> 4;

    float acc[8];
#pragma unroll
    for (int r = 0; r < 8; ++r) acc[r] = 0.f;

    for (int dt = wv; dt < NT; dt += 8) {
        const int   d   = dt * 16 + n;
        const float fcb = fc_b[d];
        float xr[8];

        for (int i = 0; i < 3; ++i) {
            v8f c8 = {};
#pragma unroll
            for (int kt = 0; kt < KT; ++kt) {
                // A fragment (16-bit 16x32 layout): row m=n, two aligned 16B chunks
                const unsigned short* arow =
                    &smA[(i * 16 + n) * PITCH + kt * 32 + half * 16];
                AB16 ua;
                ua.q[0] = *(const v4u*)(arow);        // K = base+0..7
                ua.q[1] = *(const v4u*)(arow + 8);    // K = base+8..15

                // B fragment: pre-packed in exact per-lane layout
                int bbase = ((((i * KT + kt) * NT + dt) * 32) + lane) * 2;
                AB16 ub;
                ub.q[0] = Bp[bbase];
                ub.q[1] = Bp[bbase + 1];

                c8 = __builtin_amdgcn_wmma_f32_16x16x32_bf16(
                        false, ua.v, false, ub.v, (short)0, c8, false, false);
            }
#pragma unroll
            for (int r = 0; r < 8; ++r) {
                int   mrow = r + 8 * half;
                float t2 = (c8[r] + fcb) * sMs[i * 16 + mrow] + sBs[i * 16 + mrow];
                xr[r] = (i == 0) ? t2 : xr[r] * t2;
            }
        }
#pragma unroll
        for (int r = 0; r < 8; ++r) {
            int   mrow = r + 8 * half;
            float v = xr[r] * R_emb[(size_t)sRidx[mrow] * EMB + d];
            // reduce over the 16 lanes of this half (wave32 butterflies)
            v += __shfl_xor(v, 1, 32);
            v += __shfl_xor(v, 2, 32);
            v += __shfl_xor(v, 4, 32);
            v += __shfl_xor(v, 8, 32);
            acc[r] += v;
        }
    }

    if (n == 0) {
#pragma unroll
        for (int r = 0; r < 8; ++r)
            atomicAdd(&sOut[r + 8 * half], acc[r]);   // ds_add_f32, cross-wave merge
    }
    __syncthreads();
    if (tid < 16) out[b0 + tid] = sOut[tid];
}

// ---------------------------------------------------------------------------
extern "C" void kernel_launch(void* const* d_in, const int* in_sizes, int n_in,
                              void* d_out, int out_size, void* d_ws, size_t ws_size,
                              hipStream_t stream)
{
    (void)in_sizes; (void)n_in; (void)out_size; (void)ws_size;

    const int*   r_idx = (const int*)  d_in[0];
    const int*   E_mat = (const int*)  d_in[1];
    const float* msp   = (const float*)d_in[2];
    const float* bsp   = (const float*)d_in[3];
    const float* E_emb = (const float*)d_in[4];
    const float* R_emb = (const float*)d_in[5];
    const float* fc2W  = (const float*)d_in[6];
    const float* fc2b  = (const float*)d_in[7];
    const float* fcW   = (const float*)d_in[8];
    const float* fcb   = (const float*)d_in[9];
    float* out = (float*)d_out;

    // Workspace: packed bf16 B operands, 3*7*25*32*8 dwords = 525 KB
    unsigned int* Bp = (unsigned int*)d_ws;
    const int prepN = 3 * KT * NT * 32 * 8;
    hype_prep_kernel<<<(prepN + 255) / 256, 256, 0, stream>>>(fc2W, fc2b, fcW, Bp);

    hype_main_kernel<<<BATCH / 16, 256, 0, stream>>>(
        r_idx, E_mat, msp, bsp, E_emb, R_emb, fcb, (const v4u*)d_ws, out);
}
